// ModulatedConv2d_70746701299843
// MI455X (gfx1250) — matrix operand order repaired and checked by reading, not compiled, not run on hardware
//
#include <hip/hip_runtime.h>

typedef _Float16 v16h __attribute__((ext_vector_type(16)));
typedef _Float16 v8h  __attribute__((ext_vector_type(8)));
typedef float    v8f  __attribute__((ext_vector_type(8)));

#define NB   16
#define CIN  256
#define COUT 256
#define HW   64
#define SDIM 512
#define PW   66          // 64 + 2*pad
#define ASTRIDE 40       // LDS row stride in halves (80B = 20 banks -> conflict-free b128 reads)

// ---------------------------------------------------------------------------
// Stage 0: x [B,CIN,64,64] f32  ->  xp [B,66,66,CIN] f16, zero-padded halo
// ---------------------------------------------------------------------------
__global__ void k_pad_nhwc(const float* __restrict__ x, _Float16* __restrict__ xp) {
    long idx = (long)blockIdx.x * blockDim.x + threadIdx.x;   // exact grid
    int ci = (int)(idx & 255);
    long t  = idx >> 8;
    int ix = (int)(t % PW); t /= PW;
    int iy = (int)(t % PW);
    int b  = (int)(t / PW);
    float v = 0.0f;
    if (iy >= 1 && iy <= HW && ix >= 1 && ix <= HW)
        v = x[(((long)b * CIN + ci) * HW + (iy - 1)) * HW + (ix - 1)];
    xp[idx] = (_Float16)v;
}

// ---------------------------------------------------------------------------
// Stage 1: s[b,ci] = 1 + mod_bias[ci] + lin_scale * <style[b], mod_weight[ci]>
// ---------------------------------------------------------------------------
__global__ void k_style(const float* __restrict__ style, const float* __restrict__ mw,
                        const float* __restrict__ mb, float* __restrict__ s) {
    int idx = blockIdx.x * 256 + threadIdx.x;     // 4096 threads
    int b = idx >> 8, ci = idx & 255;
    const float lin_scale = 0.044194173824159216f;  // 1/sqrt(512)
    const float* st = style + (long)b * SDIM;
    const float* w  = mw + (long)ci * SDIM;
    float acc = 0.0f;
    for (int k = 0; k < SDIM; ++k) acc += st[k] * w[k];
    s[idx] = acc * lin_scale + mb[ci] + 1.0f;
}

// ---------------------------------------------------------------------------
// Stage 2: per-(b,co) modulate + demodulate, pack f16 as wf16[b][co][kpos][ci]
// ---------------------------------------------------------------------------
__global__ void k_wmod(const float* __restrict__ weight, const float* __restrict__ s,
                       _Float16* __restrict__ wf16) {
    int b  = blockIdx.x >> 8;
    int co = blockIdx.x & 255;
    int ci = threadIdx.x;                     // 256 threads
    const float conv_scale = 0.020833333333333332f;   // 1/sqrt(256*9)
    float sv = s[b * 256 + ci] * conv_scale;
    const float* wp = weight + ((long)co * CIN + ci) * 9;
    float w9[9];
    float sum = 0.0f;
#pragma unroll
    for (int k = 0; k < 9; ++k) { w9[k] = wp[k] * sv; sum += w9[k] * w9[k]; }
    // wave32 shuffle reduce, then cross-wave via LDS
    for (int off = 16; off > 0; off >>= 1) sum += __shfl_down(sum, off, 32);
    __shared__ float red[8];
    int lane = threadIdx.x & 31, wv = threadIdx.x >> 5;
    if (lane == 0) red[wv] = sum;
    __syncthreads();
    if (wv == 0) {
        float v = (lane < 8) ? red[lane] : 0.0f;
        for (int off = 4; off > 0; off >>= 1) v += __shfl_down(v, off, 32);
        if (lane == 0) red[0] = rsqrtf(v + 1e-8f);
    }
    __syncthreads();
    float dm = red[0];
    _Float16* o = wf16 + (((long)b * COUT + co) * 9) * CIN + ci;
#pragma unroll
    for (int k = 0; k < 9; ++k) o[k * CIN] = (_Float16)(w9[k] * dm);
}

// ---------------------------------------------------------------------------
// Stage 3: implicit-GEMM conv via v_wmma_f32_16x16x32_f16.
// Block = 8 waves sharing one A panel (same b, co0) staged through LDS
// (double-buffered, 4KB/step). Each wave owns a 64(co) x 64(pixel) tile
// (one output row), i.e. 4x4 WMMA tiles, K = 9 taps * 256 ci.
// ---------------------------------------------------------------------------
__global__ __launch_bounds__(256) void k_conv_wmma(
    const _Float16* __restrict__ wf16, const _Float16* __restrict__ xp,
    const float* __restrict__ noise, const float* __restrict__ nscale,
    const float* __restrict__ bias, float* __restrict__ out) {

    __shared__ _Float16 Ash[2][64 * ASTRIDE];   // 2 x 5KB

    int tid  = threadIdx.x;
    int lane = tid & 31;
    int wv   = tid >> 5;                        // wave in block (0..7)
    int b    = blockIdx.x >> 5;                 // 32 blocks per sample
    int r    = blockIdx.x & 31;
    int co0  = (r >> 3) << 6;                   // co tile base (0,64,128,192)
    int oh   = ((r & 7) << 3) + wv;             // output row (each wave: one row)
    int llo  = lane & 15, lhi = lane >> 4;

    const _Float16* wbase = wf16 + (long)b * COUT * 9 * CIN;
    const _Float16* xbase = xp   + (long)b * PW * PW * CIN;

    // cooperative A loader: thread -> (row 0..63, 8-half chunk 0..3)
    int arow = tid >> 2;
    int acq  = (tid & 3) * 8;
    const _Float16* aml = wbase + ((long)(co0 + arow) * 9) * CIN + acq;  // + kpos*CIN + ci0
    int aoff = arow * ASTRIDE + acq;

    v8f zero = {};
    v8f acc[4][4];
#pragma unroll
    for (int i = 0; i < 4; ++i)
#pragma unroll
        for (int j = 0; j < 4; ++j) acc[i][j] = zero;

    // preload K-step 0 (kpos=0, ci0=0) into buffer 0
    {
        v8h g = *(const v8h*)aml;
        *(v8h*)&Ash[0][aoff] = g;
    }
    __syncthreads();

    for (int ks = 0; ks < 72; ++ks) {           // kpos = ks>>3, ci0 = (ks&7)*32
        int kpos = ks >> 3;
        int ci0  = (ks & 7) << 5;
        int buf  = ks & 1;

        // kick off next A panel fetch (uniform guard -> no divergence)
        v8h gnext = {};
        bool have_next = (ks + 1) < 72;
        if (have_next) {
            int kp2 = (ks + 1) >> 3;
            int c2  = ((ks + 1) & 7) << 5;
            gnext = *(const v8h*)(aml + (long)kp2 * CIN + c2);
        }

        // B fragments from global (padded NHWC): lane = pixel column,
        // lane<16 holds K 0-15, lane>=16 holds K 16-31 (contiguous ci)
        int kh = kpos / 3, kw = kpos - 3 * kh;
        const _Float16* xrow = xbase + ((long)(oh + kh) * PW + kw) * CIN + ci0 + lhi * 16;
        v16h bf[4];
#pragma unroll
        for (int j = 0; j < 4; ++j)
            bf[j] = *(const v16h*)(xrow + (long)(j * 16 + llo) * CIN);

        // A fragments from LDS: lane<16 holds K 0-15; lane>=16 K 16-31
        v16h af[4];
#pragma unroll
        for (int i = 0; i < 4; ++i) {
            const _Float16* ap = &Ash[buf][(i * 16 + llo) * ASTRIDE + lhi * 16];
            v8h lo = *(const v8h*)ap;
            v8h hi = *(const v8h*)(ap + 8);
            af[i] = __builtin_shufflevector(lo, hi,
                        0,1,2,3,4,5,6,7,8,9,10,11,12,13,14,15);
        }

#pragma unroll
        for (int i = 0; i < 4; ++i)
#pragma unroll
            for (int j = 0; j < 4; ++j)
                acc[i][j] = __builtin_amdgcn_wmma_f32_16x16x32_f16(
                    false, af[i], false, bf[j], (short)0, acc[i][j], false, false);

        if (have_next)
            *(v8h*)&Ash[buf ^ 1][aoff] = gnext;
        __syncthreads();
    }

    // Epilogue: + noise*scaler + bias, leaky_relu(0.2)
#pragma unroll
    for (int j = 0; j < 4; ++j) {
        int ow = j * 16 + llo;
        float nv = noise[((long)b * HW + oh) * HW + ow];
#pragma unroll
        for (int i = 0; i < 4; ++i) {
#pragma unroll
            for (int rr = 0; rr < 8; ++rr) {
                int co = co0 + i * 16 + lhi * 8 + rr;   // D layout: lane<16 M=rr, else M=rr+8
                float v = acc[i][j][rr] + nv * nscale[co] + bias[co];
                v = v > 0.0f ? v : 0.2f * v;
                out[(((long)b * COUT + co) * HW + oh) * HW + ow] = v;
            }
        }
    }
}

// ---------------------------------------------------------------------------
extern "C" void kernel_launch(void* const* d_in, const int* in_sizes, int n_in,
                              void* d_out, int out_size, void* d_ws, size_t ws_size,
                              hipStream_t stream) {
    const float* x      = (const float*)d_in[0];
    const float* style  = (const float*)d_in[1];
    const float* noise  = (const float*)d_in[2];
    const float* weight = (const float*)d_in[3];
    const float* mw     = (const float*)d_in[4];
    const float* mb     = (const float*)d_in[5];
    const float* nsc    = (const float*)d_in[6];
    const float* bias   = (const float*)d_in[7];
    float* out = (float*)d_out;

    // workspace layout
    char* ws = (char*)d_ws;
    float*    s    = (float*)ws;                                        // 16 KB
    _Float16* wf16 = (_Float16*)(ws + 16384);                           // 18.9 MB
    _Float16* xp   = (_Float16*)(ws + 16384 + (size_t)NB*COUT*9*CIN*2); // 35.7 MB

    const long padN = (long)NB * PW * PW * CIN;                         // 17,842,176 (mult of 256)
    k_pad_nhwc<<<(int)(padN / 256), 256, 0, stream>>>(x, xp);
    k_style  <<<16, 256, 0, stream>>>(style, mw, mb, s);
    k_wmod   <<<NB * COUT, 256, 0, stream>>>(weight, s, wf16);
    k_conv_wmma<<<512, 256, 0, stream>>>(wf16, xp, noise, nsc, bias, out);
}
